// Transformer_38319698215548
// MI455X (gfx1250) — matrix operand order repaired and checked by reading, not compiled, not run on hardware
//
#include <hip/hip_runtime.h>

// ---------------------------------------------------------------------------
// CDNA5 (gfx1250) transformer forward: bf16 WMMA GEMMs with f32 accumulation.
// Weight tiles delivered to LDS by the Tensor Data Mover (tensor_load_to_lds),
// wave32; WMMA fragment layouts per CDNA5 ISA 7.12.2, D# per ISA ch.8.
// ---------------------------------------------------------------------------

typedef __bf16 bf16;
typedef __attribute__((ext_vector_type(16))) __bf16 v16bf;
typedef __attribute__((ext_vector_type(8)))  __bf16 v8bf;
typedef __attribute__((ext_vector_type(8)))  float  v8f;
typedef __attribute__((ext_vector_type(4)))  unsigned int v4u;
typedef __attribute__((ext_vector_type(8)))  int v8i;
typedef __attribute__((ext_vector_type(4)))  int v4i;

#define BSZ 4
#define SEQ 512
#define DM  512
#define DFF 2048
#define NH  8
#define HD  64
#define MROWS (BSZ*SEQ)          // 2048
#define NEGBIG (-1.0e9f)

#define TM 128
#define TN 128
#define TK 32

// ---------------------------------------------------------------------------
// Embedding + positional encoding
// ---------------------------------------------------------------------------
__global__ __launch_bounds__(512) void embed_kernel(const int* __restrict__ tok,
                                                    const float* __restrict__ emb,
                                                    const float* __restrict__ pe,
                                                    float* __restrict__ out)
{
    int row = blockIdx.x;
    int c   = threadIdx.x;
    int t   = tok[row];
    int s   = row & (SEQ - 1);
    out[(size_t)row * DM + c] = emb[(size_t)t * DM + c] + pe[(size_t)s * DM + c];
}

// ---------------------------------------------------------------------------
// Weight prep: Wt[N][K] (bf16) = transpose(W[K][N] f32).  Block (32,8).
// Done once per weight right before its GEMM; tiles then TDM-able as-is.
// ---------------------------------------------------------------------------
__global__ __launch_bounds__(256) void transpose_cvt_kernel(const float* __restrict__ W,
                                                            __bf16* __restrict__ Wt,
                                                            int K, int N)
{
    __shared__ float t[32][33];
    int n0 = blockIdx.x * 32, k0 = blockIdx.y * 32;
    int tx = threadIdx.x, ty = threadIdx.y;
    #pragma unroll
    for (int i = 0; i < 32; i += 8)
        t[ty + i][tx] = W[(size_t)(k0 + ty + i) * N + n0 + tx];
    __syncthreads();
    #pragma unroll
    for (int i = 0; i < 32; i += 8)
        Wt[(size_t)(n0 + ty + i) * K + k0 + tx] = (__bf16)t[tx][ty + i];
}

// ---------------------------------------------------------------------------
// Fused residual + LayerNorm (D = 512), in-place safe.
// ---------------------------------------------------------------------------
__global__ __launch_bounds__(256) void ln_residual_kernel(const float* __restrict__ X,
                                                          const float* __restrict__ R,
                                                          const float* __restrict__ G,
                                                          const float* __restrict__ Bv,
                                                          float* __restrict__ Y)
{
    __shared__ float sred[8];
    __shared__ float smu, svar;
    int row = blockIdx.x, tid = threadIdx.x;
    const float* x = X + (size_t)row * DM;
    const float* r = R + (size_t)row * DM;
    float v0 = x[tid] + r[tid];
    float v1 = x[tid + 256] + r[tid + 256];

    float s = v0 + v1;
    #pragma unroll
    for (int o = 16; o > 0; o >>= 1) s += __shfl_xor(s, o, 32);
    if ((tid & 31) == 0) sred[tid >> 5] = s;
    __syncthreads();
    if (tid == 0) { float t = 0.f; for (int i = 0; i < 8; ++i) t += sred[i]; smu = t * (1.f / DM); }
    __syncthreads();
    float mu = smu;
    float d0 = v0 - mu, d1 = v1 - mu;
    float q = d0 * d0 + d1 * d1;
    #pragma unroll
    for (int o = 16; o > 0; o >>= 1) q += __shfl_xor(q, o, 32);
    __syncthreads();
    if ((tid & 31) == 0) sred[tid >> 5] = q;
    __syncthreads();
    if (tid == 0) { float t = 0.f; for (int i = 0; i < 8; ++i) t += sred[i]; svar = t * (1.f / DM); }
    __syncthreads();
    float rstd = rsqrtf(svar + 1e-5f);
    Y[(size_t)row * DM + tid]       = d0 * rstd * G[tid]       + Bv[tid];
    Y[(size_t)row * DM + tid + 256] = d1 * rstd * G[tid + 256] + Bv[tid + 256];
}

// ---------------------------------------------------------------------------
// GEMM: C[M,N] = A[M,K](f32) x W(bf16, pre-transposed [N][K]) + bias, opt ReLU.
// Block 128x128, 8 waves, wave tile 64x32. B tile (128 rows x 32 k, bf16) is
// fetched by the Tensor Data Mover straight into LDS (one tensor_load_to_lds
// per K-step, issued by wave 0, overlapped with A staging). A tile is staged
// f32->bf16 by all waves with batched loads.
// ---------------------------------------------------------------------------
__global__ __launch_bounds__(256) void gemm_bias_kernel(const float* __restrict__ A,
                                                        const __bf16* __restrict__ Wt,
                                                        const float* __restrict__ bias,
                                                        float* __restrict__ C,
                                                        int M, int K, int N, int relu)
{
    __shared__ __bf16 As[TM * TK];   // [m][k] row-major, 8 KB
    __shared__ __bf16 Bs[TN * TK];   // [n][k] row-major, 8 KB (TDM destination)

    const int tid  = threadIdx.x;
    const int wave = tid >> 5, lane = tid & 31;
    const int half = lane >> 4, lr = lane & 15;
    const int wm = (wave >> 2) * 64;
    const int wn = (wave & 3) * 32;
    const int m0 = blockIdx.y * TM;
    const int n0 = blockIdx.x * TN;

    // ---- TDM descriptor (CDNA5 ISA ch.8), invariant parts ----
    // group1: workgroup_mask=0, data_size=1(2B), tensor_dim0=K, tensor_dim1=N,
    //         tile_dim0=TK, tile_dim1=TN, tile_dim2=0, dim0_stride=K
    v8i g1;
    g1[0] = 0x00010000;                                        // data_size = 2 bytes
    g1[1] = (int)(((unsigned)K & 0xFFFFu) << 16);              // tensor_dim0[15:0]
    g1[2] = (int)((((unsigned)K) >> 16) | (((unsigned)N & 0xFFFFu) << 16));
    g1[3] = (int)((((unsigned)N) >> 16) | ((unsigned)TK << 16));
    g1[4] = TN;                                                // tile_dim1, tile_dim2=0
    g1[5] = K;                                                 // dim0_stride[31:0]
    g1[6] = 0;                                                 // dim0_stride[47:32], dim1_stride lo
    g1[7] = 0;
    v4i gz;  gz[0] = 0; gz[1] = 0; gz[2] = 0; gz[3] = 0;       // 2-D tensor: groups 2/3 unused
    v8i gz8; gz8[0]=0; gz8[1]=0; gz8[2]=0; gz8[3]=0; gz8[4]=0; gz8[5]=0; gz8[6]=0; gz8[7]=0;

    const unsigned long long gbase =
        (unsigned long long)(uintptr_t)Wt + (unsigned long long)n0 * (unsigned)K * 2ull;
    const unsigned int lds_b = (unsigned int)(uintptr_t)(&Bs[0]);

    v8f acc[4][2];
    #pragma unroll
    for (int mt = 0; mt < 4; ++mt)
        #pragma unroll
        for (int nt = 0; nt < 2; ++nt)
            #pragma unroll
            for (int i = 0; i < 8; ++i) acc[mt][nt][i] = 0.f;

    const int r  = tid >> 3;            // 0..31 (A staging row group)
    const int c4 = (tid & 7) * 4;       // 0..28

    for (int k0 = 0; k0 < K; k0 += TK) {
        // ---- issue TDM fetch of B tile (wave 0), overlapped with A staging ----
        if (wave == 0) {
            unsigned long long ga = gbase + (unsigned long long)k0 * 2ull;
            v4u g0;
            g0[0] = 1u;                                        // count = 1 descriptor
            g0[1] = lds_b;                                     // lds_addr
            g0[2] = (unsigned int)ga;                          // global_addr[31:0]
            g0[3] = (unsigned int)((ga >> 32) & 0x01FFFFFFu) | 0x80000000u; // addr[56:32] | type=2
            __builtin_amdgcn_tensor_load_to_lds(g0, g1, gz, gz, gz8, 0);
        }

        // ---- stage A tile (128 x 32 f32 -> bf16): batch loads, then stores ----
        {
            float4 fa[4];
            #pragma unroll
            for (int i = 0; i < 4; ++i)
                fa[i] = *(const float4*)(A + (size_t)(m0 + r + i * 32) * K + k0 + c4);
            if (k0 + TK < K)
                __builtin_prefetch(A + (size_t)(m0 + r) * K + k0 + TK + c4, 0, 1);
            #pragma unroll
            for (int i = 0; i < 4; ++i) {
                __bf16* dst = &As[(r + i * 32) * TK + c4];
                dst[0] = (__bf16)fa[i].x; dst[1] = (__bf16)fa[i].y;
                dst[2] = (__bf16)fa[i].z; dst[3] = (__bf16)fa[i].w;
            }
        }

        if (wave == 0) __builtin_amdgcn_s_wait_tensorcnt(0);   // B tile landed in LDS
        __syncthreads();

        // ---- fragments (ISA 7.12.2 layouts) ----
        v16bf afrag[4], bfrag[2];
        #pragma unroll
        for (int mt = 0; mt < 4; ++mt) {
            int row = wm + mt * 16 + lr;
            v8bf lo = *(const v8bf*)(&As[row * TK + half * 8]);
            v8bf hi = *(const v8bf*)(&As[row * TK + 16 + half * 8]);
            #pragma unroll
            for (int i = 0; i < 8; ++i) { afrag[mt][i] = lo[i]; afrag[mt][8 + i] = hi[i]; }
        }
        #pragma unroll
        for (int nt = 0; nt < 2; ++nt) {
            int col = wn + nt * 16 + lr;
            v8bf blo = *(const v8bf*)(&Bs[col * TK + half * 8]);
            v8bf bhi = *(const v8bf*)(&Bs[col * TK + 16 + half * 8]);
            #pragma unroll
            for (int i = 0; i < 8; ++i) { bfrag[nt][i] = blo[i]; bfrag[nt][8 + i] = bhi[i]; }
        }
        #pragma unroll
        for (int mt = 0; mt < 4; ++mt)
            #pragma unroll
            for (int nt = 0; nt < 2; ++nt)
                acc[mt][nt] = __builtin_amdgcn_wmma_f32_16x16x32_bf16(
                    false, afrag[mt], false, bfrag[nt], (short)0, acc[mt][nt], false, false);
        __syncthreads();   // ds_loads complete (waited before WMMA) -> safe to refill
    }

    // ---- epilogue ----
    #pragma unroll
    for (int mt = 0; mt < 4; ++mt)
        #pragma unroll
        for (int nt = 0; nt < 2; ++nt) {
            int n = n0 + wn + nt * 16 + lr;
            float bv = bias[n];
            #pragma unroll
            for (int rr = 0; rr < 8; ++rr) {
                int m = m0 + wm + mt * 16 + rr + half * 8;
                float v = acc[mt][nt][rr] + bv;
                if (relu) v = fmaxf(v, 0.f);
                C[(size_t)m * N + n] = v;
            }
        }
}

// ---------------------------------------------------------------------------
// Multi-head attention core (S=512, HD=64, H=8, B=4).
// grid = (B*H, 8 q-chunks); 128 threads = 4 waves, one 16-row q-tile per wave.
// ---------------------------------------------------------------------------
#define ATT_SMEM_BYTES (64*512*2 /*Vt*/ + 4*16*512*4 /*Sc*/ + 4*16*512*2 /*P*/ + 4*16*4 /*rowsum*/)

__global__ __launch_bounds__(128) void attention_kernel(const float* __restrict__ Qg,
                                                        const float* __restrict__ Kg,
                                                        const float* __restrict__ Vg,
                                                        float* __restrict__ Og,
                                                        int causal)
{
    extern __shared__ char smem[];
    __bf16* Vt     = (__bf16*)smem;                                   // [HD][SEQ]
    float*  ScBase = (float*) (smem + 64*512*2);                      // [4][16][SEQ]
    __bf16* PbBase = (__bf16*)(smem + 64*512*2 + 4*16*512*4);         // [4][16][SEQ]
    float*  rowsum = (float*) (smem + 64*512*2 + 4*16*512*4 + 4*16*512*2);

    const int bh = blockIdx.x;
    const int b = bh >> 3, h = bh & (NH - 1);
    const int tid = threadIdx.x;
    const int wave = tid >> 5, lane = tid & 31;
    const int half = lane >> 4, lr = lane & 15;

    float*  Sc = ScBase + wave * 16 * SEQ;
    __bf16* Pb = PbBase + wave * 16 * SEQ;

    // ---- stage V transposed: Vt[d][k] = V[b,k,h*64+d] ----
    {
        int d  = tid & 63;
        int k0 = tid >> 6;
        const float* vp = Vg + (size_t)b * SEQ * DM + h * HD + d;
        for (int k = k0; k < SEQ; k += 2)
            Vt[d * SEQ + k] = (__bf16)vp[(size_t)k * DM];
    }
    __syncthreads();

    const int q0 = (blockIdx.y * 4 + wave) * 16;

    // ---- Q fragments ----
    v16bf aq[2];
    {
        const float* qp = Qg + (size_t)(b * SEQ + q0 + lr) * DM + h * HD;
        #pragma unroll
        for (int dt = 0; dt < 2; ++dt) {
            const float4* p4 = (const float4*)(qp + dt * 32 + half * 8);
            float4 f0 = p4[0], f1 = p4[1], f2 = p4[4], f3 = p4[5];
            aq[dt][0]  = (__bf16)f0.x; aq[dt][1]  = (__bf16)f0.y; aq[dt][2]  = (__bf16)f0.z; aq[dt][3]  = (__bf16)f0.w;
            aq[dt][4]  = (__bf16)f1.x; aq[dt][5]  = (__bf16)f1.y; aq[dt][6]  = (__bf16)f1.z; aq[dt][7]  = (__bf16)f1.w;
            aq[dt][8]  = (__bf16)f2.x; aq[dt][9]  = (__bf16)f2.y; aq[dt][10] = (__bf16)f2.z; aq[dt][11] = (__bf16)f2.w;
            aq[dt][12] = (__bf16)f3.x; aq[dt][13] = (__bf16)f3.y; aq[dt][14] = (__bf16)f3.z; aq[dt][15] = (__bf16)f3.w;
        }
    }

    // ---- scores: S[16 x 512] = Q Kt * 0.125 (+ causal) ----
    for (int kt = 0; kt < 32; ++kt) {
        v8f s;
        #pragma unroll
        for (int i = 0; i < 8; ++i) s[i] = 0.f;
        const float* kp = Kg + (size_t)(b * SEQ + kt * 16 + lr) * DM + h * HD;
        #pragma unroll
        for (int dt = 0; dt < 2; ++dt) {
            const float4* p4 = (const float4*)(kp + dt * 32 + half * 8);
            v16bf bk;
            #pragma unroll
            for (int j = 0; j < 4; ++j) {
                float4 f = p4[j < 2 ? j : j + 2];
                bk[4*j+0] = (__bf16)f.x; bk[4*j+1] = (__bf16)f.y;
                bk[4*j+2] = (__bf16)f.z; bk[4*j+3] = (__bf16)f.w;
            }
            s = __builtin_amdgcn_wmma_f32_16x16x32_bf16(false, aq[dt], false, bk,
                                                        (short)0, s, false, false);
        }
        #pragma unroll
        for (int rr = 0; rr < 8; ++rr) {
            int m = rr + half * 8;
            int kglob = kt * 16 + lr;
            float v = s[rr] * 0.125f;
            if (causal && kglob > q0 + m) v += NEGBIG;
            Sc[m * SEQ + kglob] = v;
        }
    }
    __syncthreads();

    // ---- softmax ----
    {
        int base = lr * SEQ + half * 256;
        float mx = -3.0e38f;
        for (int k = 0; k < 256; ++k) mx = fmaxf(mx, Sc[base + k]);
        mx = fmaxf(mx, __shfl_xor(mx, 16, 32));
        float sum = 0.f;
        for (int k = 0; k < 256; ++k) {
            float e = __expf(Sc[base + k] - mx);
            sum += e;
            Pb[base + k] = (__bf16)e;
        }
        sum += __shfl_xor(sum, 16, 32);
        if (half == 0) rowsum[wave * 16 + lr] = sum;
    }
    __syncthreads();

    // ---- O[16 x 64] = P x Vt ----
    v8f oacc[4];
    #pragma unroll
    for (int nt = 0; nt < 4; ++nt)
        #pragma unroll
        for (int i = 0; i < 8; ++i) oacc[nt][i] = 0.f;

    for (int kt2 = 0; kt2 < 16; ++kt2) {
        v16bf ap;
        const __bf16* pp = Pb + lr * SEQ + kt2 * 32 + half * 8;
        v8bf lo = *(const v8bf*)pp;
        v8bf hi = *(const v8bf*)(pp + 16);
        #pragma unroll
        for (int i = 0; i < 8; ++i) { ap[i] = lo[i]; ap[8 + i] = hi[i]; }
        #pragma unroll
        for (int nt = 0; nt < 4; ++nt) {
            const __bf16* vp = Vt + (nt * 16 + lr) * SEQ + kt2 * 32 + half * 8;
            v8bf vlo = *(const v8bf*)vp;
            v8bf vhi = *(const v8bf*)(vp + 16);
            v16bf bv;
            #pragma unroll
            for (int i = 0; i < 8; ++i) { bv[i] = vlo[i]; bv[8 + i] = vhi[i]; }
            oacc[nt] = __builtin_amdgcn_wmma_f32_16x16x32_bf16(false, ap, false, bv,
                                                              (short)0, oacc[nt], false, false);
        }
    }

    #pragma unroll
    for (int nt = 0; nt < 4; ++nt)
        #pragma unroll
        for (int rr = 0; rr < 8; ++rr) {
            int m = rr + half * 8;
            float inv = 1.f / rowsum[wave * 16 + m];
            Og[(size_t)(b * SEQ + q0 + m) * DM + h * HD + nt * 16 + lr] = oacc[nt][rr] * inv;
        }
}

// ---------------------------------------------------------------------------
// Host orchestration
// ---------------------------------------------------------------------------
struct LinP { const float* w; const float* b; };
struct LnP  { const float* g; const float* b; };
struct MhaP { LinP wq, wk, wv, wo; };
struct EncL { MhaP mha; LnP ln1, ln2; LinP l1, l2; };
struct DecL { MhaP mha1, mha2; LnP ln1, ln2, ln3; LinP l1, l2; };

extern "C" void kernel_launch(void* const* d_in, const int* in_sizes, int n_in,
                              void* d_out, int out_size, void* d_ws, size_t ws_size,
                              hipStream_t stream)
{
    (void)in_sizes; (void)n_in; (void)out_size; (void)ws_size;

    const int* src = (const int*)d_in[0];
    const int* tgt = (const int*)d_in[1];
    // d_in[2..4]: all-false padding masks + causal triu (applied analytically).

    // ---- params: jax-style flatten of nested dict (sorted keys, DFS) ----
    int p = 5;
    auto nextf = [&]() { return (const float*)d_in[p++]; };
    auto rdLin = [&](LinP& L) { L.b = nextf(); L.w = nextf(); };
    auto rdLn  = [&](LnP& L)  { L.b = nextf(); L.g = nextf(); };
    auto rdMha = [&](MhaP& M) { rdLin(M.wk); rdLin(M.wo); rdLin(M.wq); rdLin(M.wv); };

    DecL dec[6];
    for (int l = 0; l < 6; ++l) {
        DecL& d = dec[l];
        rdLin(d.l1); rdLin(d.l2);
        rdLn(d.ln1); rdLn(d.ln2); rdLn(d.ln3);
        rdMha(d.mha1); rdMha(d.mha2);
    }
    EncL enc[6];
    for (int l = 0; l < 6; ++l) {
        EncL& e = enc[l];
        rdLin(e.l1); rdLin(e.l2);
        rdLn(e.ln1); rdLn(e.ln2);
        rdMha(e.mha);
    }
    LinP outp; rdLin(outp);
    const float* pe     = nextf();
    const float* srcEmb = nextf();
    const float* tgtEmb = nextf();

    // ---- workspace carve: 11*MD floats + bf16 weight scratch (max 32000x512) ----
    const size_t MD = (size_t)MROWS * DM;
    float* ws  = (float*)d_ws;
    float* X   = ws;            // encoder activations / enc_out
    float* Y   = ws + 1 * MD;   // decoder activations
    float* Qb  = ws + 2 * MD;
    float* Kb  = ws + 3 * MD;
    float* Vb  = ws + 4 * MD;
    float* ATT = ws + 5 * MD;
    float* T1  = ws + 6 * MD;
    float* HFF = ws + 7 * MD;   // [2048, 2048] = 4*MD
    __bf16* Wt = (__bf16*)(ws + 11 * MD);  // up to 32000*512 bf16 (~32.8 MB)

    auto gemm = [&](const float* A, const LinP& L, float* C, int M, int K, int N, int relu) {
        dim3 tg(N / 32, K / 32), tb(32, 8);
        transpose_cvt_kernel<<<tg, tb, 0, stream>>>(L.w, Wt, K, N);
        dim3 g(N / TN, M / TM);
        gemm_bias_kernel<<<g, 256, 0, stream>>>(A, Wt, L.b, C, M, K, N, relu);
    };
    auto attn = [&](const float* q, const float* k, const float* v, float* o, int causal) {
        dim3 g(BSZ * NH, 8);
        attention_kernel<<<g, 128, ATT_SMEM_BYTES, stream>>>(q, k, v, o, causal);
    };
    auto lnres = [&](const float* x, const float* r, const LnP& L, float* y) {
        ln_residual_kernel<<<MROWS, 256, 0, stream>>>(x, r, L.g, L.b, y);
    };

    // ================= encoder =================
    embed_kernel<<<MROWS, DM, 0, stream>>>(src, srcEmb, pe, X);
    for (int l = 0; l < 6; ++l) {
        EncL& e = enc[l];
        gemm(X, e.mha.wq, Qb, MROWS, DM, DM, 0);
        gemm(X, e.mha.wk, Kb, MROWS, DM, DM, 0);
        gemm(X, e.mha.wv, Vb, MROWS, DM, DM, 0);
        attn(Qb, Kb, Vb, ATT, /*causal=*/0);
        gemm(ATT, e.mha.wo, T1, MROWS, DM, DM, 0);
        lnres(X, T1, e.ln1, X);
        gemm(X, e.l1, HFF, MROWS, DM, DFF, /*relu=*/1);
        gemm(HFF, e.l2, T1, MROWS, DFF, DM, 0);
        lnres(X, T1, e.ln2, X);
    }

    // ================= decoder =================
    embed_kernel<<<MROWS, DM, 0, stream>>>(tgt, tgtEmb, pe, Y);
    for (int l = 0; l < 6; ++l) {
        DecL& d = dec[l];
        gemm(Y, d.mha1.wq, Qb, MROWS, DM, DM, 0);
        gemm(Y, d.mha1.wk, Kb, MROWS, DM, DM, 0);
        gemm(Y, d.mha1.wv, Vb, MROWS, DM, DM, 0);
        attn(Qb, Kb, Vb, ATT, /*causal=*/1);
        gemm(ATT, d.mha1.wo, T1, MROWS, DM, DM, 0);
        lnres(Y, T1, d.ln1, Y);
        gemm(Y, d.mha2.wq, Qb, MROWS, DM, DM, 0);
        gemm(X, d.mha2.wk, Kb, MROWS, DM, DM, 0);
        gemm(X, d.mha2.wv, Vb, MROWS, DM, DM, 0);
        attn(Qb, Kb, Vb, ATT, /*causal=*/0);
        gemm(ATT, d.mha2.wo, T1, MROWS, DM, DM, 0);
        lnres(Y, T1, d.ln2, Y);
        gemm(Y, d.l1, HFF, MROWS, DM, DFF, /*relu=*/1);
        gemm(HFF, d.l2, T1, MROWS, DFF, DM, 0);
        lnres(Y, T1, d.ln3, Y);
    }

    // ================= output projection [2048,512]x[512,32000] =================
    gemm(Y, outp, (float*)d_out, MROWS, DM, 32000, 0);
}
